// MultiHeadAttention_84327387890210
// MI455X (gfx1250) — matrix operand order repaired and checked by reading, not compiled, not run on hardware
//
#include <hip/hip_runtime.h>
#include <hip/hip_bf16.h>

// Problem constants
constexpr int Bc = 2, Sc = 2048, Dc = 1024, Hc = 16, DKc = 64, DVc = 64;

typedef __bf16 bf16;
typedef __attribute__((ext_vector_type(8)))  bf16  bf16x8;
typedef __attribute__((ext_vector_type(16))) bf16  bf16x16;
typedef __attribute__((ext_vector_type(8)))  float floatx8;

__device__ __forceinline__ bf16x16 cat16(bf16x8 lo, bf16x8 hi) {
    return __builtin_shufflevector(lo, hi, 0,1,2,3,4,5,6,7,8,9,10,11,12,13,14,15);
}

// A fragment (16x32, rows m0..m0+15, cols k0..k0+31) from row-major [.., ld] bf16.
// ISA layout: lanes 0-15 -> K {k0..k0+7, k0+16..k0+23}; lanes 16-31 -> +8.
__device__ __forceinline__ bf16x16 load_frag_a(const bf16* __restrict__ A, int ld,
                                               int m0, int k0, int lane) {
    const int l16 = lane & 15, half = lane >> 4;
    const bf16* p = A + (size_t)(m0 + l16) * ld + k0 + half * 8;
    bf16x8 lo = *(const bf16x8*)(p);
    bf16x8 hi = *(const bf16x8*)(p + 16);
    return cat16(lo, hi);
}

// B fragment (32x16, rows k0..k0+31, cols n0..n0+15) where B(k,n) = Wt[n][k]
// (Wt row-major [N, K]); lanes 0-15 -> K {k0..k0+7, k0+16..k0+23}; lanes 16-31 -> +8.
__device__ __forceinline__ bf16x16 load_frag_bt(const bf16* __restrict__ Wt, int ld,
                                                int k0, int n0, int lane) {
    const int l16 = lane & 15, half = lane >> 4;
    const bf16* p = Wt + (size_t)(n0 + l16) * ld + k0 + half * 8;
    bf16x8 lo = *(const bf16x8*)(p);
    bf16x8 hi = *(const bf16x8*)(p + 16);
    return cat16(lo, hi);
}

__device__ __forceinline__ floatx8 wmma_bf16(bf16x16 a, bf16x16 b, floatx8 c) {
    return __builtin_amdgcn_wmma_f32_16x16x32_bf16(false, a, false, b, (short)0, c,
                                                   false, false);
}

// ---------------- f32 -> bf16 conversion ----------------
__global__ void cvt_bf16_kernel(const float* __restrict__ in, bf16* __restrict__ out, int n) {
    int i = blockIdx.x * blockDim.x + threadIdx.x;
    if (i < n) out[i] = (bf16)in[i];
}

// ---------------- QKV projection GEMM ----------------
// C[M,N] = A[M,Kd] * W[N,Kd]^T ; OUT_MODE 0: write [b,h,s,dk] ; 1: write [b,h,dv,s]
template <int OUT_MODE>
__global__ void gemm_qkv_kernel(const bf16* __restrict__ A, const bf16* __restrict__ W,
                                bf16* __restrict__ Out, int Kd) {
    const int lane = threadIdx.x & 31;
    const int wid  = threadIdx.x >> 5;
    const int m0 = blockIdx.y * 64 + (wid >> 1) * 32;
    const int n0 = blockIdx.x * 64 + (wid & 1) * 32;
    floatx8 acc[2][2] = {};
    for (int k0 = 0; k0 < Kd; k0 += 32) {
        bf16x16 a0 = load_frag_a(A, Kd, m0,      k0, lane);
        bf16x16 a1 = load_frag_a(A, Kd, m0 + 16, k0, lane);
        bf16x16 b0 = load_frag_bt(W, Kd, k0, n0,      lane);
        bf16x16 b1 = load_frag_bt(W, Kd, k0, n0 + 16, lane);
        acc[0][0] = wmma_bf16(a0, b0, acc[0][0]);
        acc[0][1] = wmma_bf16(a0, b1, acc[0][1]);
        acc[1][0] = wmma_bf16(a1, b0, acc[1][0]);
        acc[1][1] = wmma_bf16(a1, b1, acc[1][1]);
    }
    const int l16 = lane & 15, half = lane >> 4;
#pragma unroll
    for (int mi = 0; mi < 2; ++mi)
#pragma unroll
        for (int ni = 0; ni < 2; ++ni)
#pragma unroll
            for (int j = 0; j < 8; ++j) {
                int m = m0 + mi * 16 + half * 8 + j;
                int n = n0 + ni * 16 + l16;
                int b = m >> 11, s = m & (Sc - 1);
                int h = n >> 6,  d = n & 63;
                size_t idx;
                if (OUT_MODE == 0)
                    idx = (((size_t)(b * Hc + h) * Sc) + s) * DKc + d;   // [b,h,s,dk]
                else
                    idx = (((size_t)(b * Hc + h) * DVc) + d) * Sc + s;   // [b,h,dv,s]
                Out[idx] = (bf16)acc[mi][ni][j];
            }
}

// ---- pass 1: per-(row, kblock) softmax partials, no raw score store ----
// Part[(bh*Sc + q)*32 + kb] = {rowmax, sum exp(x - rowmax)} over this 64-col block.
__global__ void stats_partial_kernel(const bf16* __restrict__ Qp, const bf16* __restrict__ Kp,
                                     float* __restrict__ Part) {
    const int bh  = blockIdx.z;
    const int q0b = blockIdx.y * 64;
    const int k0b = blockIdx.x * 64;
    if (k0b > q0b + 63) return;  // fully masked block
    const bf16* Q = Qp + (size_t)bh * Sc * DKc;
    const bf16* K = Kp + (size_t)bh * Sc * DKc;
    const int lane = threadIdx.x & 31;
    const int wid  = threadIdx.x >> 5;
    const int m0 = q0b + (wid >> 1) * 32;
    const int n0 = k0b + (wid & 1) * 32;
    floatx8 acc[2][2] = {};
#pragma unroll
    for (int k0 = 0; k0 < DKc; k0 += 32) {
        bf16x16 a0 = load_frag_a(Q, DKc, m0,      k0, lane);
        bf16x16 a1 = load_frag_a(Q, DKc, m0 + 16, k0, lane);
        bf16x16 b0 = load_frag_bt(K, DKc, k0, n0,      lane);
        bf16x16 b1 = load_frag_bt(K, DKc, k0, n0 + 16, lane);
        acc[0][0] = wmma_bf16(a0, b0, acc[0][0]);
        acc[0][1] = wmma_bf16(a0, b1, acc[0][1]);
        acc[1][0] = wmma_bf16(a1, b0, acc[1][0]);
        acc[1][1] = wmma_bf16(a1, b1, acc[1][1]);
    }
    __shared__ float tile[64][65];  // padded: bank = (r + c) % 64
    const int l16 = lane & 15, half = lane >> 4;
#pragma unroll
    for (int mi = 0; mi < 2; ++mi)
#pragma unroll
        for (int ni = 0; ni < 2; ++ni)
#pragma unroll
            for (int j = 0; j < 8; ++j) {
                int r = (wid >> 1) * 32 + mi * 16 + half * 8 + j;  // in-block row
                int c = (wid & 1) * 32 + ni * 16 + l16;            // in-block col
                float v = acc[mi][ni][j] * 0.125f;                 // 1/sqrt(64)
                if (k0b + c > q0b + r) v = -1e9f;
                tile[r][c] = v;
            }
    __syncthreads();
    if (threadIdx.x < 64) {
        const int r = threadIdx.x;
        float M = -3.4e38f;
#pragma unroll 8
        for (int c = 0; c < 64; ++c) M = fmaxf(M, tile[r][c]);
        float L = 0.f;
#pragma unroll 8
        for (int c = 0; c < 64; ++c) L += __expf(tile[r][c] - M);
        size_t pidx = (((size_t)bh * Sc + q0b + r) * 32 + (k0b >> 6)) * 2;
        Part[pidx]     = M;
        Part[pidx + 1] = L;
    }
}

// ---- combine block partials into per-row (max, 1/sumexp) ----
__global__ void stats_reduce_kernel(const float* __restrict__ Part, float* __restrict__ Stats) {
    int idx = blockIdx.x * blockDim.x + threadIdx.x;  // bh*Sc + q
    if (idx >= Bc * Hc * Sc) return;
    const int q  = idx & (Sc - 1);
    const int nb = (q >> 6) + 1;                      // valid causal k-blocks
    const float* p = Part + (size_t)idx * 64;
    float M = -3.4e38f;
    for (int i = 0; i < nb; ++i) M = fmaxf(M, p[2 * i]);
    float L = 0.f;
    for (int i = 0; i < nb; ++i) L += p[2 * i + 1] * __expf(p[2 * i] - M);
    Stats[2 * idx]     = M;
    Stats[2 * idx + 1] = 1.0f / L;
}

// ---- zero-fill strictly-above-diagonal 64-blocks of each attention row ----
__global__ void fill_upper_kernel(float* __restrict__ Attn) {
    const size_t row = blockIdx.x;          // bh*Sc + q
    const int q = (int)(row & (Sc - 1));
    const int kstart = ((q >> 6) + 1) << 6; // pass 2 covers k < 64*(q/64+1)
    if (kstart >= Sc) return;
    float4* p = (float4*)(Attn + row * Sc);
    const float4 z = make_float4(0.f, 0.f, 0.f, 0.f);
    for (int i = (kstart >> 2) + threadIdx.x; i < Sc / 4; i += blockDim.x) p[i] = z;
}

// ---- pass 2: recompute scores, normalize, emit softmax output, fused attn @ V ----
// One block per 64-row strip. Score tiles recomputed by WMMA (Q/K are L2-resident),
// normalized with final row stats, written to d_out, stashed as bf16 in LDS and fed
// back through LDS as WMMA-A fragments for the P @ V accumulation.
__global__ void attn_av_kernel(const bf16* __restrict__ Qp, const bf16* __restrict__ Kp,
                               const float* __restrict__ Stats, const bf16* __restrict__ Vt,
                               float* __restrict__ Attn, bf16* __restrict__ QKV) {
    const int bh = blockIdx.z;
    const int b = bh >> 4, h = bh & 15;
    const int q0b = blockIdx.y * 64;
    const int lane = threadIdx.x & 31;
    const int wid  = threadIdx.x >> 5;
    const int l16 = lane & 15, half = lane >> 4;

    const bf16* Q = Qp + (size_t)bh * Sc * DKc;
    const bf16* K = Kp + (size_t)bh * Sc * DKc;
    const bf16* V = Vt + (size_t)bh * DVc * Sc;   // [dv, s]
    float* out = Attn + (size_t)bh * Sc * Sc;

    __shared__ float sM[64];
    __shared__ float sI[64];
    constexpr int PSTR = 72;                      // bf16 row stride (conflict-free)
    __shared__ bf16 ptile[64 * PSTR];

    if (threadIdx.x < 64) {
        const float* st = Stats + 2 * ((size_t)bh * Sc + q0b + threadIdx.x);
        sM[threadIdx.x] = st[0];
        sI[threadIdx.x] = st[1];
    }
    __syncthreads();

    // scores geometry: wave quarter (rq, cq); AV geometry: wave rows rav..rav+15
    const int rq  = (wid >> 1) * 32;
    const int cq  = (wid & 1) * 32;
    const int rav = wid * 16;

    // Q fragments are k-loop invariant: hoist (2 row-frags x 2 kk-steps)
    bf16x16 qf[2][2];
#pragma unroll
    for (int mi = 0; mi < 2; ++mi)
#pragma unroll
        for (int kk = 0; kk < 2; ++kk)
            qf[mi][kk] = load_frag_a(Q, DKc, q0b + rq + mi * 16, kk * 32, lane);

    floatx8 accv[4] = {};
    const int nkb = (q0b >> 6) + 1;               // causal k-blocks for this strip
    for (int kb = 0; kb < nkb; ++kb) {
        const int k0b = kb * 64;
        // ---- score quarter: rows q0b+rq..+31, cols k0b+cq..+31 ----
        floatx8 acc[2][2] = {};
#pragma unroll
        for (int kk = 0; kk < 2; ++kk) {
            bf16x16 b0 = load_frag_bt(K, DKc, kk * 32, k0b + cq,      lane);
            bf16x16 b1 = load_frag_bt(K, DKc, kk * 32, k0b + cq + 16, lane);
            acc[0][0] = wmma_bf16(qf[0][kk], b0, acc[0][0]);
            acc[0][1] = wmma_bf16(qf[0][kk], b1, acc[0][1]);
            acc[1][0] = wmma_bf16(qf[1][kk], b0, acc[1][0]);
            acc[1][1] = wmma_bf16(qf[1][kk], b1, acc[1][1]);
        }
        // ---- normalize, emit softmax output, stash bf16 P in LDS ----
#pragma unroll
        for (int mi = 0; mi < 2; ++mi)
#pragma unroll
            for (int ni = 0; ni < 2; ++ni)
#pragma unroll
                for (int j = 0; j < 8; ++j) {
                    int r = rq + mi * 16 + half * 8 + j;
                    int c = cq + ni * 16 + l16;
                    int q = q0b + r, k = k0b + c;
                    float v = acc[mi][ni][j] * 0.125f;
                    if (k > q) v = -1e9f;
                    float p = __expf(v - sM[r]) * sI[r];
                    out[(size_t)q * Sc + k] = p;   // softmax output
                    ptile[r * PSTR + c] = (bf16)p;
                }
        __syncthreads();
        // ---- P @ V: rows rav..rav+15 x dv 0..63, K = 64 from LDS ----
#pragma unroll
        for (int kf = 0; kf < 2; ++kf) {
            const bf16* pl = &ptile[(rav + l16) * PSTR + kf * 32 + half * 8];
            bf16x8 lo = *(const bf16x8*)(pl);
            bf16x8 hi = *(const bf16x8*)(pl + 16);
            bf16x16 af = cat16(lo, hi);
#pragma unroll
            for (int ni = 0; ni < 4; ++ni) {
                bf16x16 bfr = load_frag_bt(V, Sc, k0b + kf * 32, ni * 16, lane);
                accv[ni] = wmma_bf16(af, bfr, accv[ni]);
            }
        }
        __syncthreads();  // ptile reused next iteration
    }
    // store QKV: [b, s, h*64 + dv]
#pragma unroll
    for (int ni = 0; ni < 4; ++ni)
#pragma unroll
        for (int j = 0; j < 8; ++j) {
            int q  = q0b + rav + half * 8 + j;
            int dv = ni * 16 + l16;
            QKV[((size_t)(b * Sc + q)) * (Hc * DVc) + h * DVc + dv] = (bf16)accv[ni][j];
        }
}

// ---------------- out = qkv @ Wfc^T + residual (f32) ----------------
__global__ void gemm_fc_kernel(const bf16* __restrict__ QKV, const bf16* __restrict__ Wfc,
                               const float* __restrict__ resid, float* __restrict__ OutF) {
    const int Kd = Hc * DVc;  // 1024
    const int lane = threadIdx.x & 31;
    const int wid  = threadIdx.x >> 5;
    const int m0 = blockIdx.y * 64 + (wid >> 1) * 32;
    const int n0 = blockIdx.x * 64 + (wid & 1) * 32;
    floatx8 acc[2][2] = {};
    for (int k0 = 0; k0 < Kd; k0 += 32) {
        bf16x16 a0 = load_frag_a(QKV, Kd, m0,      k0, lane);
        bf16x16 a1 = load_frag_a(QKV, Kd, m0 + 16, k0, lane);
        bf16x16 b0 = load_frag_bt(Wfc, Kd, k0, n0,      lane);
        bf16x16 b1 = load_frag_bt(Wfc, Kd, k0, n0 + 16, lane);
        acc[0][0] = wmma_bf16(a0, b0, acc[0][0]);
        acc[0][1] = wmma_bf16(a0, b1, acc[0][1]);
        acc[1][0] = wmma_bf16(a1, b0, acc[1][0]);
        acc[1][1] = wmma_bf16(a1, b1, acc[1][1]);
    }
    const int l16 = lane & 15, half = lane >> 4;
#pragma unroll
    for (int mi = 0; mi < 2; ++mi)
#pragma unroll
        for (int ni = 0; ni < 2; ++ni)
#pragma unroll
            for (int j = 0; j < 8; ++j) {
                int m = m0 + mi * 16 + half * 8 + j;
                int n = n0 + ni * 16 + l16;
                OutF[(size_t)m * Dc + n] = acc[mi][ni][j] + resid[(size_t)m * Dc + n];
            }
}

// ---------------- LayerNorm, one block per row ----------------
__global__ void ln_kernel(const float* __restrict__ X, const float* __restrict__ gamma,
                          const float* __restrict__ beta, float* __restrict__ Out) {
    const float* p = X + (size_t)blockIdx.x * Dc;
    __shared__ float r1[256];
    __shared__ float r2[256];
    float s = 0.f, s2 = 0.f;
    float v[4];
#pragma unroll
    for (int i = 0; i < 4; ++i) {
        v[i] = p[threadIdx.x + i * 256];
        s += v[i]; s2 += v[i] * v[i];
    }
    r1[threadIdx.x] = s; r2[threadIdx.x] = s2; __syncthreads();
    for (int st = 128; st > 0; st >>= 1) {
        if (threadIdx.x < st) {
            r1[threadIdx.x] += r1[threadIdx.x + st];
            r2[threadIdx.x] += r2[threadIdx.x + st];
        }
        __syncthreads();
    }
    float mu  = r1[0] * (1.0f / Dc);
    float var = r2[0] * (1.0f / Dc) - mu * mu;
    float rs  = rsqrtf(var + 1e-5f);
#pragma unroll
    for (int i = 0; i < 4; ++i) {
        int c = threadIdx.x + i * 256;
        Out[(size_t)blockIdx.x * Dc + c] = (v[i] - mu) * rs * gamma[c] + beta[c];
    }
}

extern "C" void kernel_launch(void* const* d_in, const int* in_sizes, int n_in,
                              void* d_out, int out_size, void* d_ws, size_t ws_size,
                              hipStream_t stream) {
    const float* inQ   = (const float*)d_in[0];
    const float* inK   = (const float*)d_in[1];
    const float* inV   = (const float*)d_in[2];
    // d_in[3] = attn_mask (causal, recomputed analytically)
    const float* WQ    = (const float*)d_in[4];
    const float* WK    = (const float*)d_in[5];
    const float* WV    = (const float*)d_in[6];
    const float* Wfc   = (const float*)d_in[7];
    const float* gamma = (const float*)d_in[8];
    const float* beta  = (const float*)d_in[9];

    const size_t nBSD  = (size_t)Bc * Sc * Dc;     // 4,194,304
    const size_t nW    = (size_t)Dc * Dc;          // 1,048,576
    const size_t nRows = (size_t)Bc * Hc * Sc;     // 65,536

    char* ws = (char*)d_ws;
    bf16* Xq   = (bf16*)ws; ws += nBSD * sizeof(bf16);
    bf16* Xk   = (bf16*)ws; ws += nBSD * sizeof(bf16);
    bf16* Xv   = (bf16*)ws; ws += nBSD * sizeof(bf16);
    bf16* WQb  = (bf16*)ws; ws += nW * sizeof(bf16);
    bf16* WKb  = (bf16*)ws; ws += nW * sizeof(bf16);
    bf16* WVb  = (bf16*)ws; ws += nW * sizeof(bf16);
    bf16* Wfcb = (bf16*)ws; ws += nW * sizeof(bf16);
    bf16* Qp   = (bf16*)ws; ws += nBSD * sizeof(bf16);   // [b,h,s,dk]
    bf16* Kp   = (bf16*)ws; ws += nBSD * sizeof(bf16);   // [b,h,s,dk]
    bf16* Vt   = (bf16*)ws; ws += nBSD * sizeof(bf16);   // [b,h,dv,s]
    bf16* QKVb = (bf16*)ws; ws += nBSD * sizeof(bf16);   // [b,s,h*dv]
    float* Part  = (float*)ws; ws += nRows * 32 * 2 * sizeof(float);  // 16 MB
    float* Stats = (float*)ws; ws += nRows * 2 * sizeof(float);       // 512 KB
    float* OutF  = (float*)ws;                            // [b,s,D] pre-LN, 16 MB

    float* outNorm = (float*)d_out;          // [B,S,D]
    float* outAttn = outNorm + nBSD;         // [B,H,S,S]

    const int T = 256;
    cvt_bf16_kernel<<<(int)((nBSD + T - 1) / T), T, 0, stream>>>(inQ, Xq, (int)nBSD);
    cvt_bf16_kernel<<<(int)((nBSD + T - 1) / T), T, 0, stream>>>(inK, Xk, (int)nBSD);
    cvt_bf16_kernel<<<(int)((nBSD + T - 1) / T), T, 0, stream>>>(inV, Xv, (int)nBSD);
    cvt_bf16_kernel<<<(int)((nW + T - 1) / T),   T, 0, stream>>>(WQ,  WQb, (int)nW);
    cvt_bf16_kernel<<<(int)((nW + T - 1) / T),   T, 0, stream>>>(WK,  WKb, (int)nW);
    cvt_bf16_kernel<<<(int)((nW + T - 1) / T),   T, 0, stream>>>(WV,  WVb, (int)nW);
    cvt_bf16_kernel<<<(int)((nW + T - 1) / T),   T, 0, stream>>>(Wfc, Wfcb, (int)nW);

    dim3 gproj(Dc / 64, (Bc * Sc) / 64);  // (16, 64)
    gemm_qkv_kernel<0><<<gproj, 128, 0, stream>>>(Xq, WQb, Qp, Dc);
    gemm_qkv_kernel<0><<<gproj, 128, 0, stream>>>(Xk, WKb, Kp, Dc);
    gemm_qkv_kernel<1><<<gproj, 128, 0, stream>>>(Xv, WVb, Vt, Dc);

    stats_partial_kernel<<<dim3(Sc / 64, Sc / 64, Bc * Hc), 128, 0, stream>>>(Qp, Kp, Part);
    stats_reduce_kernel<<<(int)(nRows / T), T, 0, stream>>>(Part, Stats);
    fill_upper_kernel<<<(int)nRows, T, 0, stream>>>(outAttn);
    attn_av_kernel<<<dim3(1, Sc / 64, Bc * Hc), 128, 0, stream>>>(Qp, Kp, Stats, Vt,
                                                                  outAttn, QKVb);

    gemm_fc_kernel<<<dim3(Dc / 64, (Bc * Sc) / 64), 128, 0, stream>>>(QKVb, Wfcb, inQ, OutF);
    ln_kernel<<<dim3(Bc * Sc), 256, 0, stream>>>(OutF, gamma, beta, outNorm);
}
